// SwinTransformerBlock_8332236554924
// MI455X (gfx1250) — matrix operand order repaired and checked by reading, not compiled, not run on hardware
//
#include <hip/hip_runtime.h>
#include <math.h>

typedef unsigned short u16;
typedef __attribute__((ext_vector_type(16))) __bf16 v16bf;
typedef __attribute__((ext_vector_type(8)))  float  v8f;

struct Frag32 { uint4 lo, hi; };           // 32 bytes = 16 bf16 lanes-worth
struct U16x16 { u16 v[16]; };              // 32 bytes, for building A frag from scalars

static __device__ __forceinline__ v16bf as_v16bf(const Frag32& f) {
    return __builtin_bit_cast(v16bf, f);
}
static __device__ __forceinline__ v16bf as_v16bf(const U16x16& f) {
    return __builtin_bit_cast(v16bf, f);
}
// fp32 -> bf16 round-to-nearest-even
static __device__ __forceinline__ u16 f2bf(float f) {
    union { float f; unsigned u; } x; x.f = f;
    unsigned r = x.u + 0x7FFFu + ((x.u >> 16) & 1u);
    return (u16)(r >> 16);
}

// ---------------------------------------------------------------------------
// CDNA5 async global->LDS staging (GV mode; tracked by ASYNCcnt)
// LDS address operand = flat shared pointer truncated to 32 bits
// (ISA 10.2: LDS flat mapping discards upper bits -> addr[31:0] is LDS offset)
// ---------------------------------------------------------------------------
static __device__ __forceinline__ unsigned lds_addr32(const void* p) {
    return (unsigned)(size_t)p;
}
static __device__ __forceinline__ void async_load_b128(unsigned lds_dst,
                                                       const void* gsrc) {
    asm volatile("global_load_async_to_lds_b128 %0, %1, off"
                 :: "v"(lds_dst), "v"(gsrc)
                 : "memory");
}
static __device__ __forceinline__ void wait_async0() {
    asm volatile("s_wait_asynccnt 0x0" ::: "memory");
}

// ---------------------------------------------------------------------------
// Problem constants
// ---------------------------------------------------------------------------
#define BM_  128
#define BN_  128
#define BK_  32
#define MTOT 32768        // B * H * W tokens
#define CDIM 768
#define NHEAD 12

// ---------------------------------------------------------------------------
// Weight convert + transpose: fp32 [K,N] -> bf16 [N,K]
// ---------------------------------------------------------------------------
__global__ void cvt_transpose_kernel(const float* __restrict__ W,
                                     u16* __restrict__ Wt, int K, int N) {
    int idx = blockIdx.x * 256 + threadIdx.x;
    if (idx >= K * N) return;
    int k = idx / N, n = idx - k * N;
    Wt[(size_t)n * K + k] = f2bf(W[idx]);
}

// ---------------------------------------------------------------------------
// LN1 + cyclic shift(-2,-2) + window partition, fp32 -> bf16 window order
// one wave per output (window-order) row; 8 rows per 256-thread block
// ---------------------------------------------------------------------------
__global__ __launch_bounds__(256) void ln1_shift_window_kernel(
    const float* __restrict__ x, const float* __restrict__ g,
    const float* __restrict__ b, u16* __restrict__ Xw) {
    const int lane = threadIdx.x & 31;
    const int wv   = threadIdx.x >> 5;
    const int wrow = blockIdx.x * 8 + wv;
    const int bi = wrow >> 7, rem = wrow & 127;
    const int win = rem >> 4, tok = rem & 15;
    const int wh = win >> 2, wc = win & 3;
    const int th = tok >> 2, tw = tok & 3;
    const int oh = (wh * 4 + th + 2) & 7;    // reverse the roll(-SHIFT)
    const int ow = (wc * 4 + tw + 2) & 15;
    const float* xp = x + ((size_t)bi * 128 + oh * 16 + ow) * CDIM;
    float v[24], s = 0.f, s2 = 0.f;
#pragma unroll
    for (int t = 0; t < 24; t++) {
        float a = xp[lane + t * 32]; v[t] = a; s += a; s2 += a * a;
    }
#pragma unroll
    for (int d = 1; d < 32; d <<= 1) {
        s += __shfl_xor(s, d, 32); s2 += __shfl_xor(s2, d, 32);
    }
    float mean = s * (1.f / CDIM);
    float rstd = rsqrtf(s2 * (1.f / CDIM) - mean * mean + 1e-5f);
    u16* op = Xw + (size_t)wrow * CDIM;
#pragma unroll
    for (int t = 0; t < 24; t++) {
        int c = lane + t * 32;
        op[c] = f2bf((v[t] - mean) * rstd * g[c] + b[c]);
    }
}

// ---------------------------------------------------------------------------
// Window reverse + unshift + residual (x + proj_out), then LN2
// writes xmid (fp32, token order) and y (bf16 LN2 output, token order)
// ---------------------------------------------------------------------------
__global__ __launch_bounds__(256) void unshift_resid_ln2_kernel(
    const float* __restrict__ Ow, const float* __restrict__ x,
    const float* __restrict__ g, const float* __restrict__ b,
    float* __restrict__ xmid, u16* __restrict__ y) {
    const int lane = threadIdx.x & 31;
    const int wv   = threadIdx.x >> 5;
    const int wrow = blockIdx.x * 8 + wv;
    const int bi = wrow >> 7, rem = wrow & 127;
    const int win = rem >> 4, tok = rem & 15;
    const int wh = win >> 2, wc = win & 3;
    const int th = tok >> 2, tw = tok & 3;
    const int oh = (wh * 4 + th + 2) & 7;
    const int ow = (wc * 4 + tw + 2) & 15;
    const size_t trow = (size_t)bi * 128 + oh * 16 + ow;
    const float* op = Ow + (size_t)wrow * CDIM;
    const float* xp = x + trow * CDIM;
    float* mp = xmid + trow * CDIM;
    float v[24], s = 0.f, s2 = 0.f;
#pragma unroll
    for (int t = 0; t < 24; t++) {
        int c = lane + t * 32;
        float a = xp[c] + op[c];
        mp[c] = a; v[t] = a; s += a; s2 += a * a;
    }
#pragma unroll
    for (int d = 1; d < 32; d <<= 1) {
        s += __shfl_xor(s, d, 32); s2 += __shfl_xor(s2, d, 32);
    }
    float mean = s * (1.f / CDIM);
    float rstd = rsqrtf(s2 * (1.f / CDIM) - mean * mean + 1e-5f);
    u16* yp = y + trow * CDIM;
#pragma unroll
    for (int t = 0; t < 24; t++) {
        int c = lane + t * 32;
        yp[c] = f2bf((v[t] - mean) * rstd * g[c] + b[c]);
    }
}

// ---------------------------------------------------------------------------
// Generic bf16 WMMA GEMM: C[M,N] = A[M,K](bf16) * Bt[N,K](bf16)^T + bias
// block tile 128x128, 8 waves of 32x64, BK=32.
// Tiles staged into LDS via CDNA5 async global->LDS (ASYNCcnt) path.
// ---------------------------------------------------------------------------
template <bool OUT_BF16, bool DO_GELU, bool DO_RESID>
__global__ __launch_bounds__(256) void wmma_gemm(
    const u16* __restrict__ A, const u16* __restrict__ Bt,
    const float* __restrict__ bias, const float* __restrict__ resid,
    void* __restrict__ out, int M, int N, int K) {
    __shared__ __align__(16) u16 As[BM_ * BK_];
    __shared__ __align__(16) u16 Bs[BN_ * BK_];
    const int bm = blockIdx.x * BM_;
    const int bn = blockIdx.y * BN_;
    const int tid  = threadIdx.x;
    const int lane = tid & 31;
    const int wv   = tid >> 5;
    const int wm   = (wv >> 1) * 32;   // 4 wave rows
    const int wn   = (wv & 1) * 64;    // 2 wave cols
    const int hs   = lane >> 4;        // half select (A/B K-split, C/D M-split)
    const int l16  = lane & 15;
    const int lr   = tid >> 1;         // staging row 0..127
    const int lc   = (tid & 1) * 16;   // staging col half

    v8f acc[2][4] = {};
    const u16* aG = A  + (size_t)(bm + lr) * K + lc;
    const u16* bG = Bt + (size_t)(bn + lr) * K + lc;
    const unsigned asA0 = lds_addr32(&As[lr * BK_ + lc]);
    const unsigned asA1 = lds_addr32(&As[lr * BK_ + lc + 8]);
    const unsigned asB0 = lds_addr32(&Bs[lr * BK_ + lc]);
    const unsigned asB1 = lds_addr32(&Bs[lr * BK_ + lc + 8]);

    for (int k0 = 0; k0 < K; k0 += BK_) {
        __syncthreads();                          // prior-iter LDS reads done
        async_load_b128(asA0, aG + k0);           // global -> LDS, no VGPR round-trip
        async_load_b128(asA1, aG + k0 + 8);
        async_load_b128(asB0, bG + k0);
        async_load_b128(asB1, bG + k0 + 8);
        if (k0 + BK_ < K) {                       // global_prefetch_b8 next tiles
            __builtin_prefetch(aG + k0 + BK_, 0, 1);
            __builtin_prefetch(bG + k0 + BK_, 0, 1);
        }
        wait_async0();                            // s_wait_asynccnt 0
        __syncthreads();                          // publish tile to all waves

        v16bf af[2], bw[4];
#pragma unroll
        for (int i = 0; i < 2; i++) {             // A frag: row=l16, K split by half
            const u16* p = &As[(wm + i * 16 + l16) * BK_];
            Frag32 f;
            f.lo = *(const uint4*)(p + hs * 8);
            f.hi = *(const uint4*)(p + hs * 8 + 16);
            af[i] = as_v16bf(f);
        }
#pragma unroll
        for (int j = 0; j < 4; j++) {             // B frag: col=l16, K split by half
            const u16* p = &Bs[(wn + j * 16 + l16) * BK_];
            Frag32 f;
            f.lo = *(const uint4*)(p + hs * 8);
            f.hi = *(const uint4*)(p + hs * 8 + 16);
            bw[j] = as_v16bf(f);
        }
#pragma unroll
        for (int i = 0; i < 2; i++)
#pragma unroll
            for (int j = 0; j < 4; j++)
                acc[i][j] = __builtin_amdgcn_wmma_f32_16x16x32_bf16(
                    false, af[i], false, bw[j], (short)0, acc[i][j], false, false);
    }

    // epilogue: C/D element (M = r + 8*hs, N = l16) in acc vgpr r
#pragma unroll
    for (int j = 0; j < 4; j++) {
        const int col = bn + wn + j * 16 + l16;
        const float bs = bias ? bias[col] : 0.f;
#pragma unroll
        for (int i = 0; i < 2; i++) {
#pragma unroll
            for (int r = 0; r < 8; r++) {
                const int row = bm + wm + i * 16 + hs * 8 + r;
                float v = acc[i][j][r] + bs;
                if (DO_GELU) v = 0.5f * v * (1.f + erff(v * 0.70710678118f));
                if (DO_RESID) v += resid[(size_t)row * N + col];
                if (OUT_BF16)
                    ((u16*)out)[(size_t)row * N + col] = f2bf(v);
                else
                    ((float*)out)[(size_t)row * N + col] = v;
            }
        }
    }
}

// ---------------------------------------------------------------------------
// Windowed attention: one wave per (window, head). N=16 tokens, HD=64.
// S = Q*K^T (2 WMMAs) -> +rel-bias +shift-mask -> softmax -> P*V (4 WMMAs)
// ---------------------------------------------------------------------------
__global__ __launch_bounds__(256) void attn_kernel(
    const u16* __restrict__ qkv,        // [2048*16, 2304] bf16, window order
    const float* __restrict__ rel,      // [49, 12]
    u16* __restrict__ aout) {           // [2048*16, 768] bf16
    __shared__ __align__(16) u16  Vt[8][64 * 32];   // per-wave V^T, cols 16..31 = 0 (K pad)
    __shared__ __align__(16) float Pl[8][16][16];   // per-wave probabilities
    const int lane = threadIdx.x & 31;
    const int wv   = threadIdx.x >> 5;
    const int gt   = blockIdx.x * 8 + wv;           // 0..24575
    const int w    = gt / NHEAD;
    const int h    = gt - w * NHEAD;
    const int hs   = lane >> 4;
    const int l16  = lane & 15;
    const int RS   = 3 * CDIM;                      // qkv row stride
    const u16* base = qkv + (size_t)w * 16 * RS;

    // stage V transposed into LDS: Vt[d][t], zero-pad t=16..31 (K padding)
#pragma unroll
    for (int e = 0; e < 32; e++) {
        int gi = e * 32 + lane;                     // 0..1023
        int t = gi >> 6, d = gi & 63;
        Vt[wv][d * 32 + t] = base[(size_t)t * RS + 2 * CDIM + h * 64 + d];
        Vt[wv][(gi >> 4) * 32 + 16 + (gi & 15)] = 0;
    }

    // S = Q * K^T : B columns are K rows -> contiguous loads
    v8f s = {};
#pragma unroll
    for (int kb = 0; kb < 64; kb += 32) {
        const u16* qp = base + (size_t)l16 * RS + h * 64 + kb;
        const u16* kp = base + (size_t)l16 * RS + CDIM + h * 64 + kb;
        Frag32 fq, fk;
        fq.lo = *(const uint4*)(qp + hs * 8);
        fq.hi = *(const uint4*)(qp + hs * 8 + 16);
        fk.lo = *(const uint4*)(kp + hs * 8);
        fk.hi = *(const uint4*)(kp + hs * 8 + 16);
        s = __builtin_amdgcn_wmma_f32_16x16x32_bf16(
            false, as_v16bf(fq), false, as_v16bf(fk), (short)0, s, false, false);
    }

    // bias + shift mask + row softmax (row = fixed vgpr r & half, across 16 lanes)
    const int iw = w & 7, wh = iw >> 2, wc = iw & 3;
    const int nth = l16 >> 2, ntw = l16 & 3;                 // key token coords
    const int ghk = wh * 4 + nth, gwk = wc * 4 + ntw;
    const int kcnt = (ghk < 4 ? 0 : (ghk < 6 ? 1 : 2)) * 3 +
                     (gwk < 12 ? 0 : (gwk < 14 ? 1 : 2));
#pragma unroll
    for (int r = 0; r < 8; r++) {
        const int m = r + hs * 8;                            // query token
        const int mth = m >> 2, mtw = m & 3;
        const int ghq = wh * 4 + mth, gwq = wc * 4 + mtw;
        const int qcnt = (ghq < 4 ? 0 : (ghq < 6 ? 1 : 2)) * 3 +
                         (gwq < 12 ? 0 : (gwq < 14 ? 1 : 2));
        const int ridx = (mth - nth + 3) * 7 + (mtw - ntw + 3);
        float lg = s[r] * 0.125f + rel[ridx * NHEAD + h] +
                   (qcnt != kcnt ? -100.f : 0.f);
        float mx = lg;
#pragma unroll
        for (int d = 1; d < 16; d <<= 1) mx = fmaxf(mx, __shfl_xor(mx, d, 32));
        float e = __expf(lg - mx);
        float sum = e;
#pragma unroll
        for (int d = 1; d < 16; d <<= 1) sum += __shfl_xor(sum, d, 32);
        Pl[wv][m][l16] = e / sum;
    }
    __syncthreads();

    // O = P * V : A = P (K padded 16->32 with zeros), B cols from transposed V
    U16x16 pt;
#pragma unroll
    for (int e = 0; e < 8; e++) pt.v[e] = f2bf(Pl[wv][l16][hs * 8 + e]);
#pragma unroll
    for (int e = 8; e < 16; e++) pt.v[e] = 0;
    const v16bf pa = as_v16bf(pt);

    v8f o[4] = {};
#pragma unroll
    for (int j = 0; j < 4; j++) {
        const u16* vp = &Vt[wv][(j * 16 + l16) * 32];
        Frag32 fv;
        fv.lo = *(const uint4*)(vp + hs * 8);
        fv.hi = *(const uint4*)(vp + hs * 8 + 16);
        o[j] = __builtin_amdgcn_wmma_f32_16x16x32_bf16(
            false, pa, false, as_v16bf(fv), (short)0, o[j], false, false);
    }
#pragma unroll
    for (int j = 0; j < 4; j++)
#pragma unroll
        for (int r = 0; r < 8; r++) {
            const int m = r + hs * 8;
            aout[((size_t)w * 16 + m) * CDIM + h * 64 + j * 16 + l16] = f2bf(o[j][r]);
        }
}

// ---------------------------------------------------------------------------
extern "C" void kernel_launch(void* const* d_in, const int* in_sizes, int n_in,
                              void* d_out, int out_size, void* d_ws, size_t ws_size,
                              hipStream_t stream) {
    const float* x      = (const float*)d_in[0];
    const float* ln1_g  = (const float*)d_in[1];
    const float* ln1_b  = (const float*)d_in[2];
    const float* qkv_w  = (const float*)d_in[3];
    const float* qkv_b  = (const float*)d_in[4];
    const float* proj_w = (const float*)d_in[5];
    const float* proj_b = (const float*)d_in[6];
    const float* rel    = (const float*)d_in[7];
    const float* ln2_g  = (const float*)d_in[8];
    const float* ln2_b  = (const float*)d_in[9];
    const float* fc1_w  = (const float*)d_in[10];
    const float* fc1_b  = (const float*)d_in[11];
    const float* fc2_w  = (const float*)d_in[12];
    const float* fc2_b  = (const float*)d_in[13];

    char* ws = (char*)d_ws;
    size_t off = 0;
    auto alloc = [&](size_t bytes) {
        char* p = ws + off; off += (bytes + 255) & ~(size_t)255; return p;
    };
    u16*   WtQkv  = (u16*)alloc((size_t)2304 * 768 * 2);
    u16*   WtProj = (u16*)alloc((size_t)768 * 768 * 2);
    u16*   WtFc1  = (u16*)alloc((size_t)1024 * 768 * 2);
    u16*   WtFc2  = (u16*)alloc((size_t)768 * 1024 * 2);
    u16*   Xw     = (u16*)alloc((size_t)MTOT * CDIM * 2);    // later reused as Y
    char*  big    =       alloc((size_t)MTOT * 2304 * 2);    // QKV -> Ow -> H1
    u16*   Aout   = (u16*)alloc((size_t)MTOT * CDIM * 2);
    float* Xmid   = (float*)alloc((size_t)MTOT * CDIM * 4);

    u16*   QKV = (u16*)big;          // attention input
    float* Ow  = (float*)big;        // proj output (QKV dead by then)
    u16*   H1  = (u16*)big;          // FC1 output  (Ow dead by then)
    u16*   Y   = Xw;                 // LN2 output  (Xw dead by then)

    // weight prep (bf16, [N,K] transposed)
    cvt_transpose_kernel<<<(768 * 2304 + 255) / 256, 256, 0, stream>>>(qkv_w, WtQkv, 768, 2304);
    cvt_transpose_kernel<<<(768 * 768 + 255) / 256, 256, 0, stream>>>(proj_w, WtProj, 768, 768);
    cvt_transpose_kernel<<<(768 * 1024 + 255) / 256, 256, 0, stream>>>(fc1_w, WtFc1, 768, 1024);
    cvt_transpose_kernel<<<(1024 * 768 + 255) / 256, 256, 0, stream>>>(fc2_w, WtFc2, 1024, 768);

    // LN1 + shift + window partition
    ln1_shift_window_kernel<<<MTOT / 8, 256, 0, stream>>>(x, ln1_g, ln1_b, Xw);

    // QKV: [32768,768] x [768,2304]
    wmma_gemm<true, false, false><<<dim3(MTOT / BM_, 2304 / BN_), 256, 0, stream>>>(
        Xw, WtQkv, qkv_b, nullptr, QKV, MTOT, 2304, 768);

    // attention: 2048 windows x 12 heads, 8 wave-tasks per block
    attn_kernel<<<(2048 * NHEAD) / 8, 256, 0, stream>>>(QKV, rel, Aout);

    // proj: [32768,768] x [768,768] (+bias) -> fp32 window order
    wmma_gemm<false, false, false><<<dim3(MTOT / BM_, CDIM / BN_), 256, 0, stream>>>(
        Aout, WtProj, proj_b, nullptr, Ow, MTOT, CDIM, CDIM);

    // window reverse + unshift + residual + LN2
    unshift_resid_ln2_kernel<<<MTOT / 8, 256, 0, stream>>>(Ow, x, ln2_g, ln2_b, Xmid, Y);

    // FC1 + GELU: [32768,768] x [768,1024]
    wmma_gemm<true, true, false><<<dim3(MTOT / BM_, 1024 / BN_), 256, 0, stream>>>(
        Y, WtFc1, fc1_b, nullptr, H1, MTOT, 1024, 768);

    // FC2 + bias + residual -> d_out (fp32)
    wmma_gemm<false, false, true><<<dim3(MTOT / BM_, CDIM / BN_), 256, 0, stream>>>(
        H1, WtFc2, fc2_b, Xmid, (float*)d_out, MTOT, CDIM, 1024);
}
